// GCN2_21792664060284
// MI455X (gfx1250) — matrix operand order repaired and checked by reading, not compiled, not run on hardware
//
#include <hip/hip_runtime.h>
#include <hip/hip_bf16.h>

typedef _Float16 half8 __attribute__((ext_vector_type(8)));
typedef _Float16 v16h  __attribute__((ext_vector_type(16)));
typedef float    v8f   __attribute__((ext_vector_type(8)));

#define NS 115
#define NQ 23
#define GS 115
#define GQ 23
#define NCLS 23
#define DD 1024
#define ES 4096
#define EQ 1024

// ---------------------------------------------------------------------------
// small utility kernels
// ---------------------------------------------------------------------------
__global__ void k_zero_f32(float* p, int n) {
    int i = blockIdx.x * blockDim.x + threadIdx.x;
    if (i < n) p[i] = 0.f;
}
__global__ void k_cast_f32_f16(const float* __restrict__ s, _Float16* __restrict__ d, int n) {
    int i = blockIdx.x * blockDim.x + threadIdx.x;
    if (i < n) d[i] = (_Float16)s[i];
}
__global__ void k_cast_f16_f32(const _Float16* __restrict__ s, float* __restrict__ d, int n) {
    int i = blockIdx.x * blockDim.x + threadIdx.x;
    if (i < n) d[i] = (float)s[i];
}
// conv1 weights [64][27] -> f16 [64][32] (K padded with zeros)
__global__ void k_pack_w1(const float* __restrict__ s, _Float16* __restrict__ d) {
    int i = blockIdx.x * blockDim.x + threadIdx.x;
    if (i >= 64 * 32) return;
    int co = i >> 5, k = i & 31;
    d[i] = (k < 27) ? (_Float16)s[co * 27 + k] : (_Float16)0.f;
}
// dst[n*K+k] = (f16) src[k*N+n]   (pack GEMM B as B^T row-major)
__global__ void k_transpose_cast(const float* __restrict__ s, _Float16* __restrict__ d, int K, int N) {
    int i = blockIdx.x * blockDim.x + threadIdx.x;
    if (i >= K * N) return;
    int n = i / K, k = i - n * K;
    d[i] = (_Float16)s[k * N + n];
}
__global__ void k_count_edges(const int* __restrict__ ei, int E, float* __restrict__ cnt) {
    int e = blockIdx.x * blockDim.x + threadIdx.x;
    if (e < E) atomicAdd(&cnt[ei[E + e]], 1.f);
}
__global__ void k_scatter_add(const float* __restrict__ x, const int* __restrict__ ei, int E,
                              float* __restrict__ acc) {
    int i = blockIdx.x * blockDim.x + threadIdx.x;
    if (i >= E * DD) return;
    int e = i >> 10, d = i & (DD - 1);
    atomicAdd(&acc[(size_t)ei[E + e] * DD + d], x[(size_t)ei[e] * DD + d]);
}
// out = (acc / max(cnt,1)) * h    elementwise over N*D
__global__ void k_agg_mul(const float* __restrict__ acc, const float* __restrict__ cnt,
                          const float* __restrict__ h, float* __restrict__ o, int n) {
    int i = blockIdx.x * blockDim.x + threadIdx.x;
    if (i >= n) return;
    float c = cnt[i >> 10];
    o[i] = (acc[i] / (c > 1.f ? c : 1.f)) * h[i];
}
// pad-cast rows: d[Mpad][K], rows >= M zero-filled
__global__ void k_pad_cast(const float* __restrict__ s, _Float16* __restrict__ d,
                           int M, int Mpad, int K) {
    int i = blockIdx.x * blockDim.x + threadIdx.x;
    if (i >= Mpad * K) return;
    int r = i / K;
    d[i] = (r < M) ? (_Float16)s[i] : (_Float16)0.f;
}
// concat [h1|h2] -> f16 [Mpad][2048]
__global__ void k_concat_cast(const float* __restrict__ a, const float* __restrict__ b,
                              _Float16* __restrict__ d, int M, int Mpad) {
    int i = blockIdx.x * blockDim.x + threadIdx.x;
    if (i >= Mpad * 2 * DD) return;
    int r = i / (2 * DD), c = i - r * 2 * DD;
    float v = 0.f;
    if (r < M) v = (c < DD) ? a[r * DD + c] : b[r * DD + c - DD];
    d[i] = (_Float16)v;
}
// 2x2 maxpool, f16 in [C][2Ho][2Wo] -> f16 out [C][Ho][Wo]
__global__ void k_maxpool(const _Float16* __restrict__ in, _Float16* __restrict__ out,
                          int C, int Ho, int Wo) {
    int i = blockIdx.x * blockDim.x + threadIdx.x;
    if (i >= C * Ho * Wo) return;
    int c = i / (Ho * Wo), rem = i - c * Ho * Wo;
    int oy = rem / Wo, ox = rem - oy * Wo;
    int H = Ho * 2, W = Wo * 2;
    const _Float16* p = in + ((size_t)c * H + 2 * oy) * W + 2 * ox;
    float v0 = (float)p[0], v1 = (float)p[1], v2 = (float)p[W], v3 = (float)p[W + 1];
    float m = v0 > v1 ? v0 : v1;
    float m2 = v2 > v3 ? v2 : v3;
    out[i] = (_Float16)(m > m2 ? m : m2);
}

// ---------------------------------------------------------------------------
// implicit-GEMM 3x3 SAME conv with fused bias+ReLU, WMMA f16->f32.
// in  : f16 [CIN][H][W];  w16 : f16 [64][KPAD], K = ci*9+ky*3+kx (zero padded)
// out : f16 [64][H][W]
// block = 128 threads = 4 waves; each wave owns a 16-output-channel tile of the
// same 16-pixel row segment. LDS stages:
//   (1) halo [ci][3][20] (cx <-> ix = x0-2+cx): interior blocks use per-lane
//       GLOBAL_LOAD_ASYNC_TO_LDS_B32 (ASYNCcnt) -- no VGPR round trip; boundary
//       blocks use a scalar zero-filled path. Uniform per block.
//   (2) im2col built directly in WMMA B^T layout bt[pixel][K], so the unrolled
//       compute loop is branch-free: 2x16B global A loads + 32B LDS B + v_wmma.
// ---------------------------------------------------------------------------
template <int CIN>
__global__ __launch_bounds__(128)
void k_conv3x3_wmma(const _Float16* __restrict__ in, const _Float16* __restrict__ w16,
                    const float* __restrict__ bias, _Float16* __restrict__ out,
                    int H, int W) {
    constexpr int KREAL = CIN * 9;
    constexpr int KPAD  = (KREAL + 31) & ~31;
    constexpr int KQ    = KPAD / 32;
    __shared__ _Float16 halo[CIN * 60];     // [ci][ry 0..2][cx 0..19]
    __shared__ _Float16 bt[16 * KPAD];      // [pixel 0..15][K]

    const int x0 = blockIdx.x * 16;
    const int y  = blockIdx.y;
    const int tid = threadIdx.x;

    // warm L2 for the next output row's inputs (emits global_prefetch_b8)
    if (tid < CIN) {
        int py = (y + 2 < H) ? y + 2 : H - 1;
        __builtin_prefetch(in + ((size_t)tid * H + py) * W + x0, 0, 1);
    }

    // stage 1: halo into LDS
    const bool interior = (x0 >= 2) && (x0 + 17 < W) && (y >= 1) && (y + 1 < H);
    if (interior) {
        // fully in-bounds: async DMA global -> LDS in dword chunks (ASYNCcnt)
        for (int d = tid; d < CIN * 30; d += 128) {
            int ci = d / 30, rem = d - ci * 30;
            int ry = rem / 10, cq = rem - ry * 10;
            const _Float16* gp = in + ((size_t)ci * H + (y + ry - 1)) * W + (x0 - 2) + 2 * cq;
            unsigned ldsoff = (unsigned)(size_t)(const void*)&halo[d * 2];
            unsigned long long ga = (unsigned long long)(size_t)(const void*)gp;
            asm volatile("global_load_async_to_lds_b32 %0, %1, off"
                         :: "v"(ldsoff), "v"(ga) : "memory");
        }
        asm volatile("s_wait_asynccnt 0" ::: "memory");
    } else {
        for (int e = tid; e < CIN * 60; e += 128) {
            int ci = e / 60, rem = e - ci * 60;
            int ry = rem / 20, cx = rem - ry * 20;
            int iy = y + ry - 1, ix = x0 + cx - 2;
            _Float16 v = (_Float16)0.f;
            if (iy >= 0 && iy < H && ix >= 0 && ix < W)
                v = in[((size_t)ci * H + iy) * W + ix];
            halo[e] = v;
        }
    }
    __syncthreads();

    // stage 2: build im2col in B^T layout; index math once per K column
    for (int e = tid; e < KPAD; e += 128) {
        _Float16 v[16];
        if (KPAD == KREAL || e < KREAL) {
            int ci = e / 9, r = e - 9 * ci;
            int ky = r / 3, kx = r - 3 * ky;
            const _Float16* src = &halo[ci * 60 + ky * 20 + kx + 1];  // p=0 -> ix = x0+kx-1
#pragma unroll
            for (int p = 0; p < 16; ++p) v[p] = src[p];
        } else {
#pragma unroll
            for (int p = 0; p < 16; ++p) v[p] = (_Float16)0.f;
        }
#pragma unroll
        for (int p = 0; p < 16; ++p) bt[p * KPAD + e] = v[p];
    }
    __syncthreads();

    const int wave = tid >> 5;
    const int lane = tid & 31;
    const bool lo  = lane < 16;
    const int  col = lane & 15;          // pixel within tile (B column / D column)
    const int  coBase = wave * 16;

    // A: lanes 0-15 hold K {0..15}, lanes 16-31 hold {16..31}
    const _Float16* wrow  = w16 + (size_t)(coBase + col) * KPAD;
    const _Float16* arow0 = wrow + (lo ? 0 : 16);
    const _Float16* arow1 = wrow + (lo ? 8 : 24);
    // B: lane holds pixel `col`, K = kk*32 + (lo?0:16) + e  (contiguous 32B)
    const _Float16* brow  = &bt[col * KPAD + (lo ? 0 : 16)];

    v8f c = {};
#pragma unroll
    for (int kk = 0; kk < KQ; ++kk) {
        half8 a0 = *(const half8*)(arow0 + kk * 32);
        half8 a1 = *(const half8*)(arow1 + kk * 32);
        v16h A;
#pragma unroll
        for (int i = 0; i < 8; ++i) { A[i] = a0[i]; A[8 + i] = a1[i]; }
        v16h B = *(const v16h*)(brow + kk * 32);
        c = __builtin_amdgcn_wmma_f32_16x16x32_f16(false, A, false, B, (short)0, c,
                                                   false, false);
    }
    // epilogue: D VGPR r -> M = r (lanes 0-15) or 8+r (lanes 16-31), N = col
    const int x = x0 + col;
    if (x < W) {
#pragma unroll
        for (int r = 0; r < 8; ++r) {
            int co = coBase + (lo ? r : 8 + r);
            float v = c[r] + bias[co];
            v = v > 0.f ? v : 0.f;
            out[((size_t)co * H + y) * W + x] = (_Float16)v;
        }
    }
}

// ---------------------------------------------------------------------------
// dense GEMM: C[M][N] = A[Mpad][K] (f16) x B (via BT f16 [N][K]) + bias, f32 out
// block = 128 threads = 4 waves; wave handles one 16-wide N tile
// ---------------------------------------------------------------------------
__global__ __launch_bounds__(128)
void k_gemm_wmma(const _Float16* __restrict__ A, const _Float16* __restrict__ BT,
                 const float* __restrict__ bias, float* __restrict__ C,
                 int M, int K, int N) {
    const int tid = threadIdx.x;
    const int wave = tid >> 5, lane = tid & 31;
    const bool lo = lane < 16;
    const int l15 = lane & 15;
    const int nBase = (blockIdx.x * 4 + wave) * 16;
    const int mBase = blockIdx.y * 16;
    if (nBase >= N) return;

    const _Float16* arow = A + (size_t)(mBase + l15) * K;
    const _Float16* brow = BT + (size_t)(nBase + l15) * K;
    v8f c = {};
#pragma unroll 4
    for (int kk = 0; kk < K; kk += 32) {
        half8 a0 = *(const half8*)(arow + kk + (lo ? 0 : 16));
        half8 a1 = *(const half8*)(arow + kk + (lo ? 8 : 24));
        v16h Af;
#pragma unroll
        for (int i = 0; i < 8; ++i) { Af[i] = a0[i]; Af[8 + i] = a1[i]; }
        v16h Bf = *(const v16h*)(brow + kk + (lo ? 0 : 16));
        c = __builtin_amdgcn_wmma_f32_16x16x32_f16(false, Af, false, Bf, (short)0, c,
                                                   false, false);
    }
    const int n = nBase + l15;
    const float bv = bias ? bias[n] : 0.f;
#pragma unroll
    for (int r = 0; r < 8; ++r) {
        int mm = mBase + (lo ? r : 8 + r);
        if (mm < M) C[(size_t)mm * N + n] = c[r] + bv;
    }
}

// ---------------------------------------------------------------------------
// fused distance term:
// bsum[q][g] = sum_d exp(-(0.25*center[g/5]+0.5*sx[g]-qx[q])^2)
//            + (sum_d a - D*log(sum_d exp(a))),  a = exp(-(sx[g]-qx[q])^2)
// (the seg-means over support_batch/query_batch are identity: 1 node/graph)
// ---------------------------------------------------------------------------
__global__ __launch_bounds__(256)
void k_pair_b(const float* __restrict__ sx, const float* __restrict__ qx,
              const float* __restrict__ center, float* __restrict__ bsum) {
    const int g = blockIdx.x;     // support graph 0..114
    const int q = blockIdx.y;     // query graph   0..22
    const float* s  = sx + (size_t)g * DD;
    const float* qv = qx + (size_t)q * DD;
    const float* cc = center + (size_t)(g / 5) * DD;
    float sa = 0.f, se = 0.f, rb = 0.f;
    for (int d = threadIdx.x; d < DD; d += 256) {
        float ds = s[d] - qv[d];
        float a = __expf(-(ds * ds));
        sa += a;
        se += __expf(a);
        float sc = 0.25f * cc[d] + 0.5f * s[d] - qv[d];
        rb += __expf(-(sc * sc));
    }
    __shared__ float r0[256], r1[256], r2[256];
    int t = threadIdx.x;
    r0[t] = sa; r1[t] = se; r2[t] = rb;
    __syncthreads();
    for (int st = 128; st > 0; st >>= 1) {
        if (t < st) { r0[t] += r0[t + st]; r1[t] += r1[t + st]; r2[t] += r2[t + st]; }
        __syncthreads();
    }
    if (t == 0)
        bsum[q * GS + g] = r2[0] + (r0[0] - (float)DD * __logf(r1[0]));
}

// dis[q][c] = mean over 5 class members of log_softmax_g(bsum[q][:])
__global__ void k_dis(const float* __restrict__ bsum, float* __restrict__ dis) {
    int q = threadIdx.x;
    if (q >= GQ) return;
    const float* row = bsum + q * GS;
    float mx = row[0];
    for (int g = 1; g < GS; ++g) mx = row[g] > mx ? row[g] : mx;
    float se = 0.f;
    for (int g = 0; g < GS; ++g) se += __expf(row[g] - mx);
    float lse = mx + __logf(se);
    for (int c = 0; c < NCLS; ++c) {
        float s = 0.f;
        for (int j = 0; j < 5; ++j) s += row[5 * c + j] - lse;
        dis[q * NCLS + c] = s * 0.2f;
    }
}

// center_new[c][d] = 0.1 * sum_{j<5} sx[5c+j][d] + 0.25*center[c][d]
__global__ void k_center_new(const float* __restrict__ sx, const float* __restrict__ center,
                             float* __restrict__ out) {
    int i = blockIdx.x * blockDim.x + threadIdx.x;
    if (i >= NCLS * DD) return;
    int c = i >> 10, d = i & (DD - 1);
    float s = 0.f;
    for (int j = 0; j < 5; ++j) s += sx[(size_t)(5 * c + j) * DD + d];
    out[i] = 0.1f * s + 0.25f * center[i];
}

__global__ void k_loss_acc(const float* __restrict__ dis, const int* __restrict__ qy,
                           float* __restrict__ out) {
    if (threadIdx.x != 0 || blockIdx.x != 0) return;
    float loss = 0.f, acc = 0.f;
    for (int q = 0; q < GQ; ++q) {
        const float* r = dis + q * NCLS;
        float mx = r[0]; int am = 0;
        for (int c = 1; c < NCLS; ++c) if (r[c] > mx) { mx = r[c]; am = c; }
        float se = 0.f;
        for (int c = 0; c < NCLS; ++c) se += __expf(r[c] - mx);
        float lse = mx + __logf(se);
        int y = qy[q];
        loss += -(r[y] - lse);
        acc += (am == y) ? 1.f : 0.f;
    }
    out[0] = loss;
    out[1] = acc;
}

// ---------------------------------------------------------------------------
// host orchestration
// ---------------------------------------------------------------------------
static inline size_t alignUp256(size_t x) { return (x + 255) & ~(size_t)255; }

extern "C" void kernel_launch(void* const* d_in, const int* in_sizes, int n_in,
                              void* d_out, int out_size, void* d_ws, size_t ws_size,
                              hipStream_t stream) {
    (void)in_sizes; (void)n_in; (void)out_size; (void)ws_size;

    const float* support_x = (const float*)d_in[0];
    const int*   support_ei = (const int*)d_in[1];
    const int*   query_y_p = (const int*)d_in[7];
    const float* query_x = (const float*)d_in[4];
    const int*   query_ei = (const int*)d_in[5];
    const float* center = (const float*)d_in[8];
    const float* cw1 = (const float*)d_in[9];
    const float* cb1 = (const float*)d_in[10];
    const float* cw_rest = (const float*)d_in[11];
    const float* cb_rest = (const float*)d_in[12];
    const float* lin2_w = (const float*)d_in[13];
    const float* lin2_b = (const float*)d_in[14];
    const float* mlp_w = (const float*)d_in[15];
    const float* mlp_b = (const float*)d_in[16];
    float* out = (float*)d_out;

    // carve workspace
    char* p = (char*)d_ws;
    auto carve = [&](size_t bytes) -> void* { void* r = (void*)p; p += alignUp256(bytes); return r; };
    _Float16* w1_16   = (_Float16*)carve(64 * 32 * 2);
    _Float16* wl_16   = (_Float16*)carve(5 * 64 * 576 * 2);
    _Float16* lin2T   = (_Float16*)carve((size_t)DD * DD * 2);
    _Float16* mlpT    = (_Float16*)carve((size_t)DD * 2 * DD * 2);
    _Float16* img16   = (_Float16*)carve((size_t)3 * 65536 * 2);
    _Float16* bufA    = (_Float16*)carve((size_t)64 * 65536 * 2);
    _Float16* bufB    = (_Float16*)carve((size_t)64 * 16384 * 2);
    float*    h1buf   = (float*)carve((size_t)128 * DD * 4);
    float*    accbuf  = (float*)carve((size_t)128 * DD * 4);
    float*    cntbuf  = (float*)carve(128 * 4);
    float*    sxbuf   = (float*)carve((size_t)128 * DD * 4);
    _Float16* a16buf  = (_Float16*)carve((size_t)128 * DD * 2);
    float*    h2buf   = (float*)carve((size_t)128 * DD * 4);
    float*    hbbuf   = (float*)carve((size_t)128 * DD * 4);
    _Float16* cat16   = (_Float16*)carve((size_t)128 * 2 * DD * 2);
    float*    sfeat   = (float*)carve((size_t)NS * DD * 4);
    float*    qfeat   = (float*)carve((size_t)NQ * DD * 4);
    float*    bsum    = (float*)carve((size_t)GQ * GS * 4);
    float*    disbuf  = (float*)carve((size_t)GQ * NCLS * 4);

    const int T = 256;
    auto blocks = [](int n, int t) { return (n + t - 1) / t; };

    // ---- pack weights (every call; deterministic) ----
    k_pack_w1<<<blocks(64 * 32, T), T, 0, stream>>>(cw1, w1_16);
    k_cast_f32_f16<<<blocks(5 * 64 * 576, T), T, 0, stream>>>(cw_rest, wl_16, 5 * 64 * 576);
    k_transpose_cast<<<blocks(DD * DD, T), T, 0, stream>>>(lin2_w, lin2T, DD, DD);
    k_transpose_cast<<<blocks(2 * DD * DD, T), T, 0, stream>>>(mlp_w, mlpT, 2 * DD, DD);

    // ---- per-set pipeline ----
    auto run_set = [&](const float* imgs, int N, const int* ei, int E, float* feat) {
        const int Mpad = (N + 15) & ~15;
        // CNN encoder per image (ping-pong bufA/bufB per image; stream is in-order)
        for (int img = 0; img < N; ++img) {
            const float* x = imgs + (size_t)img * 3 * 65536;
            k_cast_f32_f16<<<blocks(3 * 65536, T), T, 0, stream>>>(x, img16, 3 * 65536);
            // conv1: 3 -> 64 @ 256x256
            k_conv3x3_wmma<3><<<dim3(16, 256), 128, 0, stream>>>(img16, w1_16, cb1, bufA,
                                                                 256, 256);
            k_maxpool<<<blocks(64 * 128 * 128, T), T, 0, stream>>>(bufA, bufB, 64, 128, 128);
            // conv2..6: 64 -> 64
            for (int l = 0; l < 5; ++l) {
                int H = 128 >> l;
                k_conv3x3_wmma<64><<<dim3((H + 15) / 16, H), 128, 0, stream>>>(
                    bufB, wl_16 + (size_t)l * 64 * 576, cb_rest + l * 64, bufA, H, H);
                int Ho = H >> 1;
                k_maxpool<<<blocks(64 * Ho * Ho, T), T, 0, stream>>>(bufA, bufB, 64, Ho, Ho);
            }
            // final pooled [64][4][4] == flatten order of reference
            k_cast_f16_f32<<<blocks(DD, T), T, 0, stream>>>(bufB, h1buf + (size_t)img * DD, DD);
        }
        // conv-GCN layer 1: sx = agg_mean(h1) * h1
        k_zero_f32<<<blocks(N * DD, T), T, 0, stream>>>(accbuf, N * DD);
        k_zero_f32<<<blocks(N, T), T, 0, stream>>>(cntbuf, N);
        k_count_edges<<<blocks(E, T), T, 0, stream>>>(ei, E, cntbuf);
        k_scatter_add<<<blocks(E * DD, T), T, 0, stream>>>(h1buf, ei, E, accbuf);
        k_agg_mul<<<blocks(N * DD, T), T, 0, stream>>>(accbuf, cntbuf, h1buf, sxbuf, N * DD);
        // h2 = sx @ lin2_w + b
        k_pad_cast<<<blocks(Mpad * DD, T), T, 0, stream>>>(sxbuf, a16buf, N, Mpad, DD);
        k_gemm_wmma<<<dim3(DD / 64, Mpad / 16), 128, 0, stream>>>(a16buf, lin2T, lin2_b,
                                                                  h2buf, N, DD, DD);
        // conv-GCN layer 2: hb = agg_mean(h2) * h2
        k_zero_f32<<<blocks(N * DD, T), T, 0, stream>>>(accbuf, N * DD);
        k_zero_f32<<<blocks(N, T), T, 0, stream>>>(cntbuf, N);
        k_count_edges<<<blocks(E, T), T, 0, stream>>>(ei, E, cntbuf);
        k_scatter_add<<<blocks(E * DD, T), T, 0, stream>>>(h2buf, ei, E, accbuf);
        k_agg_mul<<<blocks(N * DD, T), T, 0, stream>>>(accbuf, cntbuf, h2buf, hbbuf, N * DD);
        // feat = concat(h1, hb) @ mlp_w + b
        k_concat_cast<<<blocks(Mpad * 2 * DD, T), T, 0, stream>>>(h1buf, hbbuf, cat16, N, Mpad);
        k_gemm_wmma<<<dim3(DD / 64, Mpad / 16), 128, 0, stream>>>(cat16, mlpT, mlp_b,
                                                                  feat, N, 2 * DD, DD);
    };

    run_set(support_x, NS, support_ei, ES, sfeat);
    run_set(query_x, NQ, query_ei, EQ, qfeat);

    // ---- distance head ----
    k_pair_b<<<dim3(GS, GQ), 256, 0, stream>>>(sfeat, qfeat, center, bsum);
    k_dis<<<1, 32, 0, stream>>>(bsum, disbuf);
    k_center_new<<<blocks(NCLS * DD, T), T, 0, stream>>>(sfeat, center, out + 2);
    k_loss_acc<<<1, 32, 0, stream>>>(disbuf, query_y_p, out);
}
